// WS_79388175499822
// MI455X (gfx1250) — matrix operand order repaired and checked by reading, not compiled, not run on hardware
//
#include <hip/hip_runtime.h>
#include <hip/hip_bf16.h>
#include <stdint.h>

// Vector types matching the gfx1250 WMMA builtin signatures.
typedef __attribute__((ext_vector_type(16))) __bf16 v16bf;
typedef __attribute__((ext_vector_type(8)))  float  v8f;
typedef __attribute__((ext_vector_type(8)))  float  vf8;

#define TN 4  // N-tiles (16 cols) per wave
#define MT 2  // M-tiles (16 rows) per wave -> 32x64 C block per wave

#define GLOBAL_AS __attribute__((address_space(1)))

// Force global (addrspace 1) memory ops so the backend emits
// global_load_b128 / global_store_b32 instead of flat_* (flat ticks both
// LOADcnt and DScnt and contends on the LDS path).
__device__ __forceinline__ v16bf ldg_bf16x16(const __bf16* p) {
    return *(const GLOBAL_AS v16bf*)(uintptr_t)p;
}
__device__ __forceinline__ vf8 ldg_f32x8(const float* p) {
    return *(const GLOBAL_AS vf8*)(uintptr_t)p;
}
__device__ __forceinline__ float ldg_f32(const float* p) {
    return *(const GLOBAL_AS float*)(uintptr_t)p;
}
__device__ __forceinline__ void stg_f32(float* p, float v) {
    *(GLOBAL_AS float*)(uintptr_t)p = v;
}

// ---------------- Phase 0: zero the f32 segment accumulator ----------------
__global__ void zero_f4(float4* __restrict__ p, long long n4) {
    long long i = (long long)blockIdx.x * blockDim.x + threadIdx.x;
    if (i < n4) p[i] = make_float4(0.f, 0.f, 0.f, 0.f);
}

// ---------------- Phase 0b: split lin_w into bf16 hi/lo --------------------
__global__ void split_bf16(const float* __restrict__ src, __bf16* __restrict__ hi,
                           __bf16* __restrict__ lo, int n) {
    int i = blockIdx.x * blockDim.x + threadIdx.x;
    if (i < n) {
        float x = src[i];
        __bf16 h = (__bf16)x;            // RNE to bf16
        hi[i] = h;
        lo[i] = (__bf16)(x - (float)h);  // residual term
    }
}

// ---------------- Phase 1: scaled scatter-add (bandwidth bound) ------------
// One block per row; 256 threads x float2 = 512 elements. One no-return
// global_atomic_add_f32 per output element; destination is L2-resident.
__global__ void scatter_rows(const float* __restrict__ h, const float* __restrict__ w,
                             const long long* __restrict__ idx, float* __restrict__ agg,
                             int in_dim, int n_nodes) {
    int row = blockIdx.x;
    float s = w[row % n_nodes];                 // tiled per-row scale
    long long seg = idx[row];
    const float* src = h + (size_t)row * in_dim;
    float* dst = agg + (size_t)seg * in_dim;
    int t = threadIdx.x * 2;
    float2 v = *(const float2*)(src + t);
    GLOBAL_AS float* d = (GLOBAL_AS float*)(uintptr_t)(dst + t);
    __hip_atomic_fetch_add(d,     s * v.x, __ATOMIC_RELAXED, __HIP_MEMORY_SCOPE_AGENT);
    __hip_atomic_fetch_add(d + 1, s * v.y, __ATOMIC_RELAXED, __HIP_MEMORY_SCOPE_AGENT);
}

// ---------------- Phase 2: split-bf16 WMMA GEMM ----------------------------
// C[s, o] = sum_k agg[s,k] * lin_w[o,k] + b[o]
// Split-bf16 3-product scheme: x = hi + lo; A*B ~= Ahi*Bhi + Ahi*Blo + Alo*Bhi
// (dropped lo*lo term ~2^-16 relative) -> fp32-grade accuracy at bf16 rate.
// Each wave computes MT=2 row tiles x TN=4 col tiles (32x64 C block), so every
// B fragment feeds 2 A strips (halves L2 B traffic, 24 WMMAs per k-step).
__global__ void __launch_bounds__(256)
gemm_bf16x3(const float* __restrict__ agg, const __bf16* __restrict__ whi,
            const __bf16* __restrict__ wlo, const float* __restrict__ bias,
            float* __restrict__ out, int K, int outd, int ngroups) {
    const int wave  = threadIdx.x >> 5;
    const int lane  = threadIdx.x & 31;
    const int gtile = blockIdx.x * (blockDim.x >> 5) + wave;
    const int tn0   = (gtile % ngroups) * TN;   // first N tile (16 cols each)
    const int tm0   = (gtile / ngroups) * MT;   // first M tile (16 rows each)
    const int l16   = lane & 15;
    const int half  = lane >> 4;

    v8f acc[MT][TN] = {};

    // A 16-bit layout: lane holds row tile*16 + l16;
    // lanes 0-15 take K {0..15}, lanes 16-31 take K {16..31}.
    const float* arow[MT];
#pragma unroll
    for (int m = 0; m < MT; ++m)
        arow[m] = agg + (size_t)((tm0 + m) * 16 + l16) * K + half * 16;

    // B 32x16 layout: lane n = l16 reads 16 contiguous k at k0 + half*16.
    const __bf16* bhp[TN];
    const __bf16* blp[TN];
#pragma unroll
    for (int t = 0; t < TN; ++t) {
        size_t boff = (size_t)((tn0 + t) * 16 + l16) * K + half * 16;
        bhp[t] = whi + boff;
        blp[t] = wlo + boff;
    }

    for (int k0 = 0; k0 < K; k0 += 32) {
        v16bf ahi[MT], alo[MT];
#pragma unroll
        for (int m = 0; m < MT; ++m) {
            vf8 c0 = ldg_f32x8(arow[m]);        // K chunk 0
            vf8 c1 = ldg_f32x8(arow[m] + 8);    // K chunk 1
            arow[m] += 32;
#pragma unroll
            for (int i = 0; i < 8; ++i) {
                float x = c0[i];
                __bf16 hb = (__bf16)x;
                ahi[m][i] = hb;
                alo[m][i] = (__bf16)(x - (float)hb);
                float y = c1[i];
                __bf16 hb2 = (__bf16)y;
                ahi[m][i + 8] = hb2;
                alo[m][i + 8] = (__bf16)(y - (float)hb2);
            }
        }
#pragma unroll
        for (int t = 0; t < TN; ++t) {
            v16bf bhi = ldg_bf16x16(bhp[t]);
            v16bf blo = ldg_bf16x16(blp[t]);
            bhp[t] += 32;
            blp[t] += 32;
#pragma unroll
            for (int m = 0; m < MT; ++m) {
                acc[m][t] = __builtin_amdgcn_wmma_f32_16x16x32_bf16(
                    false, ahi[m], false, bhi, (short)0, acc[m][t], false, false);
                acc[m][t] = __builtin_amdgcn_wmma_f32_16x16x32_bf16(
                    false, ahi[m], false, blo, (short)0, acc[m][t], false, false);
                acc[m][t] = __builtin_amdgcn_wmma_f32_16x16x32_bf16(
                    false, alo[m], false, bhi, (short)0, acc[m][t], false, false);
            }
        }
    }

    // C/D layout: VGPR r -> M = r + half*8, N = l16
#pragma unroll
    for (int t = 0; t < TN; ++t) {
        int col = (tn0 + t) * 16 + l16;
        float b = ldg_f32(bias + col);
#pragma unroll
        for (int m = 0; m < MT; ++m) {
#pragma unroll
            for (int r = 0; r < 8; ++r) {
                int row = (tm0 + m) * 16 + r + half * 8;
                stg_f32(out + (size_t)row * outd + col, acc[m][t][r] + b);
            }
        }
    }
}

extern "C" void kernel_launch(void* const* d_in, const int* in_sizes, int n_in,
                              void* d_out, int out_size, void* d_ws, size_t ws_size,
                              hipStream_t stream) {
    const float*     h     = (const float*)d_in[0];
    const float*     w     = (const float*)d_in[1];
    const float*     lin_w = (const float*)d_in[2];
    const float*     lin_b = (const float*)d_in[3];
    const long long* idx   = (const long long*)d_in[4];

    const int outd    = in_sizes[3];            // 512
    const int in_dim  = in_sizes[2] / outd;     // 512
    const int n_rows  = in_sizes[0] / in_dim;   // 409600
    const int n_nodes = in_sizes[1];            // 51200
    const int segs    = out_size / outd;        // 65536

    // Workspace layout: agg f32 [segs, in_dim] | w_hi bf16 | w_lo bf16
    float*  agg = (float*)d_ws;
    size_t agg_elems = (size_t)segs * in_dim;
    __bf16* whi = (__bf16*)((char*)d_ws + agg_elems * sizeof(float));
    __bf16* wlo = whi + (size_t)outd * in_dim;

    // 0) zero agg
    long long n4 = (long long)(agg_elems / 4);
    zero_f4<<<(int)((n4 + 255) / 256), 256, 0, stream>>>((float4*)agg, n4);

    // 0b) split lin_w into hi/lo bf16 (1 MB, reused by every M strip)
    int wn = outd * in_dim;
    split_bf16<<<(wn + 255) / 256, 256, 0, stream>>>(lin_w, whi, wlo, wn);

    // 1) scaled scatter-add
    scatter_rows<<<n_rows, in_dim / 2, 0, stream>>>(h, w, idx, agg, in_dim, n_nodes);

    // 2) WMMA GEMM: (segs/32) M-groups x (outd/64) N-groups, 8 waves/block
    int ngroups = outd / (16 * TN);                                // 8
    long long waves = (long long)(segs / (16 * MT)) * ngroups;     // 16384
    gemm_bf16x3<<<(int)(waves / 8), 256, 0, stream>>>(agg, whi, wlo, lin_b,
                                                      (float*)d_out, in_dim, outd, ngroups);
}
